// LoRAAttnProcessor_30399778521251
// MI455X (gfx1250) — matrix operand order repaired and checked by reading, not compiled, not run on hardware
//
#include <hip/hip_runtime.h>
#include <hip/hip_bf16.h>

// ---------------------------------------------------------------------------
// LoRA attention for MI455X (gfx1250).
//  - LoRA folded into K/V weights:  k + (k@A)@B == x @ (Wk + Wk A B)
//  - bf16 WMMA (v_wmma_f32_16x16x32_bf16) for all GEMMs + attention
//  - V written pre-transposed by the QKV GEMM -> barrier-free flash attention
//  - softmax scale folded into Q at the GEMM store
//  - speculative global_prefetch of next K/V chunk, split across waves
// ---------------------------------------------------------------------------

#define HIDDEN 1280
#define HEADS  8
#define RANK   4
#define BATCH  2
#define SEQ    4096
#define DH     (HIDDEN / HEADS)   // 160
#define MROWS  (BATCH * SEQ)      // 8192
#define NQKV   (3 * HIDDEN)       // 3840
#define QKLD   (2 * HIDDEN)       // 2560 (Q at col 0, K at col 1280)

typedef __bf16 bf16;
typedef __bf16 v16bf __attribute__((ext_vector_type(16)));
typedef float  v8f   __attribute__((ext_vector_type(8)));

union AFrag { v16bf v; float4 f[2]; };

__device__ __forceinline__ v8f zero_v8f() {
  v8f z = {0.f, 0.f, 0.f, 0.f, 0.f, 0.f, 0.f, 0.f};
  return z;
}

// A-matrix fragment (16x32 bf16, row-major source, leading dim `ld` elements).
// Lanes 0-15 -> rows M=0..15, K = k0+0..15 contiguous;
// lanes 16-31 -> same rows, K = k0+16..31 contiguous (matches load_b_frag).
__device__ __forceinline__ v16bf load_a_frag(const bf16* base, int ld,
                                             int row0, int k0, int lane) {
  int r  = row0 + (lane & 15);
  int kk = k0 + ((lane >> 4) << 4);
  const bf16* p = base + (size_t)r * ld + kk;
  AFrag u;
  u.f[0] = *(const float4*)(p);        // K kk .. kk+7
  u.f[1] = *(const float4*)(p + 8);    // K kk+8 .. kk+15
  return u.v;
}

// B-matrix fragment (32x16 bf16). Source is B^T row-major: row n of `bt` is
// column n of B, contiguous in K. Order used here: lane holds column N = lane%16,
// K = k0 + (lane/16)*16 .. +15, contiguous.
__device__ __forceinline__ v16bf load_b_frag(const bf16* bt, int ld,
                                             int col0, int k0, int lane) {
  int n  = col0 + (lane & 15);
  int kk = k0 + ((lane >> 4) << 4);
  const bf16* p = bt + (size_t)n * ld + kk;
  AFrag u;
  u.f[0] = *(const float4*)(p);        // K kk .. kk+7
  u.f[1] = *(const float4*)(p + 8);    // K kk+8 .. kk+15
  return u.v;
}

__device__ __forceinline__ v8f wmma_bf16(v16bf a, v16bf b, v8f c) {
  return __builtin_amdgcn_wmma_f32_16x16x32_bf16(
      /*neg_a=*/false, a, /*neg_b=*/false, b,
      /*c_mod=*/(short)0, c, /*reuse_a=*/false, /*reuse_b=*/false);
}

// --------------------------- small prep kernels ----------------------------

// WkA[i,r] = sum_o Wk[i,o] * A[o,r]   (and same for V). 1280x4 each.
__global__ void lora_wa_kernel(const float* __restrict__ w_k,
                               const float* __restrict__ w_v,
                               const float* __restrict__ a_k,
                               const float* __restrict__ a_v,
                               float* __restrict__ wka,
                               float* __restrict__ wva) {
  int idx = blockIdx.x * blockDim.x + threadIdx.x;
  if (idx >= 2 * HIDDEN * RANK) return;
  int which = idx / (HIDDEN * RANK);
  int rem = idx - which * (HIDDEN * RANK);
  int i = rem / RANK, r = rem - (rem / RANK) * RANK;
  const float* W = which ? w_v : w_k;
  const float* A = which ? a_v : a_k;
  float s = 0.f;
  for (int o = 0; o < HIDDEN; ++o)
    s += W[(size_t)i * HIDDEN + o] * A[(size_t)o * RANK + r];
  (which ? wva : wka)[rem] = s;
}

__global__ void pack_x_kernel(const float* __restrict__ x,
                              bf16* __restrict__ xb, int n) {
  int idx = blockIdx.x * blockDim.x + threadIdx.x;
  if (idx < n) xb[idx] = (bf16)x[idx];
}

// Fused, transposed weight pack: wt[o, i] for o in [0,3840):
//   o <  1280 : Wq[i,o]
//   o < 2560  : Wk[i,oo] + sum_r WkA[i,r] * Bk[r,oo]   (LoRA fold)
//   else      : Wv[i,oo] + sum_r WvA[i,r] * Bv[r,oo]
__global__ void pack_wqkv_kernel(const float* __restrict__ w_q,
                                 const float* __restrict__ w_k,
                                 const float* __restrict__ w_v,
                                 const float* __restrict__ bk,
                                 const float* __restrict__ bv,
                                 const float* __restrict__ wka,
                                 const float* __restrict__ wva,
                                 bf16* __restrict__ wt) {
  int idx = blockIdx.x * blockDim.x + threadIdx.x;
  if (idx >= NQKV * HIDDEN) return;
  int o = idx / HIDDEN;
  int i = idx - o * HIDDEN;
  float val;
  if (o < HIDDEN) {
    val = w_q[(size_t)i * HIDDEN + o];
  } else if (o < 2 * HIDDEN) {
    int oo = o - HIDDEN;
    val = w_k[(size_t)i * HIDDEN + oo];
#pragma unroll
    for (int r = 0; r < RANK; ++r)
      val += wka[i * RANK + r] * bk[r * HIDDEN + oo];
  } else {
    int oo = o - 2 * HIDDEN;
    val = w_v[(size_t)i * HIDDEN + oo];
#pragma unroll
    for (int r = 0; r < RANK; ++r)
      val += wva[i * RANK + r] * bv[r * HIDDEN + oo];
  }
  wt[idx] = (bf16)val;
}

__global__ void pack_wout_kernel(const float* __restrict__ w_o,
                                 bf16* __restrict__ wt) {
  int idx = blockIdx.x * blockDim.x + threadIdx.x;
  if (idx >= HIDDEN * HIDDEN) return;
  int o = idx / HIDDEN;
  int i = idx - o * HIDDEN;
  wt[idx] = (bf16)w_o[(size_t)i * HIDDEN + o];
}

// ------------------------------- QKV GEMM ----------------------------------
// [8192, 3840] = xb[8192, 1280] @ W (W transposed in memory).
// Block = 8 waves, wave tile 32x64 (2 A-frags x 4 B-frags, 8 accumulators),
// block tile 128x128. Q section is pre-scaled by 1/sqrt(dh); V section is
// written TRANSPOSED into vt[b, dh_col, s] so attention's PV B-fragments are
// contiguous global loads (no LDS transpose, no block barriers downstream).
__global__ __launch_bounds__(256) void gemm_qkv_kernel(
    const bf16* __restrict__ xb, const bf16* __restrict__ wt,
    bf16* __restrict__ qk, bf16* __restrict__ vt) {
  int lane = threadIdx.x & 31;
  int w = threadIdx.x >> 5;
  int m0 = blockIdx.x * 128 + (w & 3) * 32;
  int n0 = blockIdx.y * 128 + (w >> 2) * 64;
  const float qscale = 0.07905694150420949f;  // 1/sqrt(160)

  v8f acc[2][4];
#pragma unroll
  for (int mi = 0; mi < 2; ++mi)
#pragma unroll
    for (int j = 0; j < 4; ++j) acc[mi][j] = zero_v8f();

  for (int k0 = 0; k0 < HIDDEN; k0 += 32) {
    v16bf a0 = load_a_frag(xb, HIDDEN, m0, k0, lane);
    v16bf a1 = load_a_frag(xb, HIDDEN, m0 + 16, k0, lane);
#pragma unroll
    for (int j = 0; j < 4; ++j) {
      v16bf b = load_b_frag(wt, HIDDEN, n0 + j * 16, k0, lane);
      acc[0][j] = wmma_bf16(a0, b, acc[0][j]);
      acc[1][j] = wmma_bf16(a1, b, acc[1][j]);
    }
  }

#pragma unroll
  for (int mi = 0; mi < 2; ++mi) {
#pragma unroll
    for (int j = 0; j < 4; ++j) {
      int col = n0 + j * 16 + (lane & 15);
      int r0 = m0 + mi * 16 + ((lane >> 4) << 3);
      if (col < HIDDEN) {              // Q (pre-scaled)
#pragma unroll
        for (int i = 0; i < 8; ++i)
          qk[(size_t)(r0 + i) * QKLD + col] = (bf16)(acc[mi][j][i] * qscale);
      } else if (col < 2 * HIDDEN) {   // K
#pragma unroll
        for (int i = 0; i < 8; ++i)
          qk[(size_t)(r0 + i) * QKLD + col] = (bf16)acc[mi][j][i];
      } else {                         // V -> transposed vt[b, c, s]
        int c = col - 2 * HIDDEN;
#pragma unroll
        for (int i = 0; i < 8; ++i) {
          int r = r0 + i;
          int bb = r >> 12;            // r / SEQ
          int s = r & (SEQ - 1);
          vt[((size_t)bb * HIDDEN + c) * SEQ + s] = (bf16)acc[mi][j][i];
        }
      }
    }
  }
}

// ----------------------------- flash attention -----------------------------
// Block = 4 waves x 16 query rows (64 q / block), chunk = 64 K/V columns.
// Q pre-scaled; K read row-major (B-frag of K^T = contiguous K rows);
// V read from the pre-transposed vt buffer. P re-fragmented through
// wave-private LDS -> no block-wide barriers at all.
__global__ __launch_bounds__(128) void attn_kernel(
    const bf16* __restrict__ qk, const bf16* __restrict__ vt,
    bf16* __restrict__ attnout) {
  __shared__ bf16 ldsP[4][16 * 64];

  int tid = threadIdx.x;
  int lane = tid & 31, w = tid >> 5;
  int b = blockIdx.z, h = blockIdx.y;
  int q0 = blockIdx.x * 64;
  const bf16* Qm = qk + (size_t)b * SEQ * QKLD + h * DH;
  const bf16* Km = Qm + HIDDEN;
  const bf16* vtH = vt + ((size_t)b * HIDDEN + h * DH) * SEQ;
  int qrow0 = q0 + w * 16;

  v16bf qf[5];
#pragma unroll
  for (int kk = 0; kk < 5; ++kk)
    qf[kk] = load_a_frag(Qm, QKLD, qrow0, kk * 32, lane);

  v8f o[10];
#pragma unroll
  for (int t = 0; t < 10; ++t) o[t] = zero_v8f();
  float m_i[8], l_i[8];
#pragma unroll
  for (int i = 0; i < 8; ++i) { m_i[i] = -1e30f; l_i[i] = 0.f; }

  for (int c0 = 0; c0 < SEQ; c0 += 64) {
    // Speculative prefetch of the NEXT chunk, split across waves
    // (wave-uniform branches; global_prefetch_b8 pulls whole cachelines).
    if (c0 + 64 < SEQ) {
      if (w == 0) {       // 160 VT lines of 128B each: 5 x 32 lanes
#pragma unroll
        for (int j = 0; j < 5; ++j)
          __builtin_prefetch(vtH + (size_t)(j * 32 + lane) * SEQ + c0 + 64, 0, 1);
      } else if (w == 1) {  // 64 K rows of 320B each: 2 x 32 row starts
        __builtin_prefetch(Km + (size_t)(c0 + 64 + lane) * QKLD, 0, 1);
        __builtin_prefetch(Km + (size_t)(c0 + 96 + lane) * QKLD, 0, 1);
      }
    }

    // scores S = Qs(16x160) @ K^T(160x64): 4 accumulators of 16x16
    v8f s[4];
#pragma unroll
    for (int j = 0; j < 4; ++j) s[j] = zero_v8f();
#pragma unroll
    for (int kk = 0; kk < 5; ++kk) {
      v16bf a = qf[kk];
#pragma unroll
      for (int j = 0; j < 4; ++j) {
        v16bf bfr = load_b_frag(Km, QKLD, c0 + j * 16, kk * 32, lane);
        s[j] = wmma_bf16(a, bfr, s[j]);
      }
    }

    // online softmax; C-layout: lane holds col N=lane%16, row i+8*(lane/16)
    int qb = ((lane >> 4) << 3);
#pragma unroll
    for (int i = 0; i < 8; ++i) {
      float t = fmaxf(fmaxf(s[0][i], s[1][i]), fmaxf(s[2][i], s[3][i]));
#pragma unroll
      for (int off = 1; off < 16; off <<= 1)
        t = fmaxf(t, __shfl_xor(t, off, 32));
      float mn = fmaxf(m_i[i], t);
      float corr = __expf(m_i[i] - mn);
      int q = i + qb;
      float ps = 0.f;
#pragma unroll
      for (int j = 0; j < 4; ++j) {
        float p = __expf(s[j][i] - mn);
        ldsP[w][q * 64 + j * 16 + (lane & 15)] = (bf16)p;
        ps += p;
      }
#pragma unroll
      for (int off = 1; off < 16; off <<= 1)
        ps += __shfl_xor(ps, off, 32);
      l_i[i] = l_i[i] * corr + ps;
      m_i[i] = mn;
#pragma unroll
      for (int t2 = 0; t2 < 10; ++t2) o[t2][i] *= corr;
    }

    // O += P(16x64) @ V(64x160); P from wave-private LDS (A-frags),
    // V columns contiguous in vt (B-frags straight from global/L2).
    v16bf pa0 = load_a_frag(ldsP[w], 64, 0, 0, lane);
    v16bf pa1 = load_a_frag(ldsP[w], 64, 0, 32, lane);
#pragma unroll
    for (int t2 = 0; t2 < 10; ++t2) {
      v16bf vb0 = load_b_frag(vtH, SEQ, t2 * 16, c0, lane);
      o[t2] = wmma_bf16(pa0, vb0, o[t2]);
      v16bf vb1 = load_b_frag(vtH, SEQ, t2 * 16, c0 + 32, lane);
      o[t2] = wmma_bf16(pa1, vb1, o[t2]);
    }
  }

#pragma unroll
  for (int i = 0; i < 8; ++i) {
    float inv = 1.f / l_i[i];
#pragma unroll
    for (int t2 = 0; t2 < 10; ++t2) o[t2][i] *= inv;
  }
#pragma unroll
  for (int t2 = 0; t2 < 10; ++t2) {
    int col = h * DH + t2 * 16 + (lane & 15);
    size_t r0 = (size_t)b * SEQ + qrow0 + ((lane >> 4) << 3);
#pragma unroll
    for (int i = 0; i < 8; ++i)
      attnout[(r0 + i) * HIDDEN + col] = (bf16)o[t2][i];
  }
}

// ----------------------------- output GEMM ---------------------------------
__global__ __launch_bounds__(256) void gemm_out_kernel(
    const bf16* __restrict__ ab, const bf16* __restrict__ wt,
    const float* __restrict__ bias, float* __restrict__ out) {
  int lane = threadIdx.x & 31;
  int w = threadIdx.x >> 5;
  int m0 = blockIdx.x * 128 + (w & 3) * 32;
  int n0 = blockIdx.y * 128 + (w >> 2) * 64;
  v8f acc[2][4];
#pragma unroll
  for (int mi = 0; mi < 2; ++mi)
#pragma unroll
    for (int j = 0; j < 4; ++j) acc[mi][j] = zero_v8f();

  for (int k0 = 0; k0 < HIDDEN; k0 += 32) {
    v16bf a0 = load_a_frag(ab, HIDDEN, m0, k0, lane);
    v16bf a1 = load_a_frag(ab, HIDDEN, m0 + 16, k0, lane);
#pragma unroll
    for (int j = 0; j < 4; ++j) {
      v16bf b = load_b_frag(wt, HIDDEN, n0 + j * 16, k0, lane);
      acc[0][j] = wmma_bf16(a0, b, acc[0][j]);
      acc[1][j] = wmma_bf16(a1, b, acc[1][j]);
    }
  }
#pragma unroll
  for (int mi = 0; mi < 2; ++mi) {
#pragma unroll
    for (int j = 0; j < 4; ++j) {
      int col = n0 + j * 16 + (lane & 15);
      float bc = bias[col];
      int r0 = m0 + mi * 16 + ((lane >> 4) << 3);
#pragma unroll
      for (int i = 0; i < 8; ++i)
        out[(size_t)(r0 + i) * HIDDEN + col] = acc[mi][j][i] + bc;
    }
  }
}

// ------------------------------ launch -------------------------------------
extern "C" void kernel_launch(void* const* d_in, const int* in_sizes, int n_in,
                              void* d_out, int out_size, void* d_ws,
                              size_t ws_size, hipStream_t stream) {
  (void)in_sizes; (void)n_in; (void)out_size; (void)ws_size;
  const float* x   = (const float*)d_in[0];
  const float* w_q = (const float*)d_in[1];
  const float* w_k = (const float*)d_in[2];
  const float* w_v = (const float*)d_in[3];
  const float* a_k = (const float*)d_in[4];
  const float* b_k = (const float*)d_in[5];
  const float* a_v = (const float*)d_in[6];
  const float* b_v = (const float*)d_in[7];
  const float* w_o = (const float*)d_in[8];
  const float* b_o = (const float*)d_in[9];
  float* out = (float*)d_out;

  char* ws = (char*)d_ws;
  size_t off = 0;
  auto wsalloc = [&](size_t bytes) -> void* {
    void* p = ws + off;
    off = (off + bytes + 255) & ~(size_t)255;
    return p;
  };
  bf16* xb    = (bf16*)wsalloc((size_t)MROWS * HIDDEN * 2);    // 20 MB
  bf16* wqkvt = (bf16*)wsalloc((size_t)NQKV * HIDDEN * 2);     // 9.8 MB
  bf16* woutt = (bf16*)wsalloc((size_t)HIDDEN * HIDDEN * 2);   // 3.3 MB
  bf16* qk    = (bf16*)wsalloc((size_t)MROWS * QKLD * 2);      // 41.9 MB
  bf16* vt    = (bf16*)wsalloc((size_t)BATCH * HIDDEN * SEQ * 2); // 21 MB
  bf16* aout  = (bf16*)wsalloc((size_t)MROWS * HIDDEN * 2);    // 20 MB
  float* wka  = (float*)wsalloc((size_t)HIDDEN * RANK * 4);
  float* wva  = (float*)wsalloc((size_t)HIDDEN * RANK * 4);

  lora_wa_kernel<<<(2 * HIDDEN * RANK + 255) / 256, 256, 0, stream>>>(
      w_k, w_v, a_k, a_v, wka, wva);
  int nx = MROWS * HIDDEN;
  pack_x_kernel<<<(nx + 255) / 256, 256, 0, stream>>>(x, xb, nx);
  pack_wqkv_kernel<<<(NQKV * HIDDEN + 255) / 256, 256, 0, stream>>>(
      w_q, w_k, w_v, b_k, b_v, wka, wva, wqkvt);
  pack_wout_kernel<<<(HIDDEN * HIDDEN + 255) / 256, 256, 0, stream>>>(w_o, woutt);

  gemm_qkv_kernel<<<dim3(MROWS / 128, NQKV / 128), 256, 0, stream>>>(
      xb, wqkvt, qk, vt);
  attn_kernel<<<dim3(SEQ / 64, HEADS, BATCH), 128, 0, stream>>>(qk, vt, aout);
  gemm_out_kernel<<<dim3(MROWS / 128, HIDDEN / 128), 256, 0, stream>>>(
      aout, woutt, b_o, out);
}
